// SEALSAGE_53420803228460
// MI455X (gfx1250) — compile-verified, matrix operand order checked by reading, not run on hardware
//
#include <hip/hip_runtime.h>
#include <hip/hip_bf16.h>

typedef __attribute__((ext_vector_type(16))) _Float16 v16h;
typedef __attribute__((ext_vector_type(8)))  float    v8f;

#define HDIM 128

// ---------------------------------------------------------------- utilities
__global__ void zero_kernel(float* __restrict__ p, int n) {
    int i = blockIdx.x * blockDim.x + threadIdx.x;
    if (i < n) p[i] = 0.0f;
}

__global__ void f32_to_f16_kernel(const float* __restrict__ src,
                                  _Float16* __restrict__ dst, int n) {
    int i = blockIdx.x * blockDim.x + threadIdx.x;
    if (i < n) dst[i] = (_Float16)src[i];
}

// h0[i, 0:128] = z_table[z[i]]; h0[i,128:256] = x[i]   (float4 granularity)
__global__ void embed_kernel(const int* __restrict__ z,
                             const float* __restrict__ x,
                             const float* __restrict__ z_table,
                             float* __restrict__ h0, int N) {
    int i = blockIdx.x * blockDim.x + threadIdx.x;   // over N*64 quads
    if (i >= N * 64) return;
    int node = i >> 6;
    int c4 = (i & 63) * 4;
    float4 v;
    if (c4 < HDIM) v = *(const float4*)&z_table[(size_t)z[node] * HDIM + c4];
    else           v = *(const float4*)&x[(size_t)node * HDIM + (c4 - HDIM)];
    *(float4*)&h0[(size_t)node * 256 + c4] = v;
}

__global__ void degree_kernel(const int* __restrict__ dst,
                              float* __restrict__ deg, int E) {
    int e = blockIdx.x * blockDim.x + threadIdx.x;
    if (e < E) atomicAdd(&deg[dst[e]], 1.0f);
}

// agg[dst] += h[src]   (d = 256 or 128, float4 per thread, 4 atomics)
__global__ void scatter_add_kernel(const float* __restrict__ h,
                                   const int* __restrict__ src,
                                   const int* __restrict__ dst,
                                   float* __restrict__ agg,
                                   int E, int dquads) {
    int i = blockIdx.x * blockDim.x + threadIdx.x;
    if (i >= E * dquads) return;
    int e  = i / dquads;
    int c4 = (i - e * dquads) * 4;
    int d  = dquads * 4;
    int s = src[e], t = dst[e];
    float4 v = *(const float4*)&h[(size_t)s * d + c4];
    float* a = &agg[(size_t)t * d + c4];
    atomicAdd(a + 0, v.x);
    atomicAdd(a + 1, v.y);
    atomicAdd(a + 2, v.z);
    atomicAdd(a + 3, v.w);
}

// first occurrence of each graph id in sorted batch
__global__ void centers_kernel(const int* __restrict__ batch,
                               int* __restrict__ centers, int N) {
    int i = blockIdx.x * blockDim.x + threadIdx.x;
    if (i >= N) return;
    int b = batch[i];
    if (i == 0 || batch[i - 1] != b) centers[b] = i;
}

// ---------------------------------------------------------------- WMMA GEMM
// out[16 x 128 tile] = (agg/deg) @ Wl^T + h @ Wr^T + bias  (optional ReLU)
// One wave per 16-node tile; 8 waves / block share LDS-staged 128x32 W slices.
template <int K, bool RELU>
__global__ __launch_bounds__(256) void sage_gemm_kernel(
    const float* __restrict__ agg, const float* __restrict__ deg,
    const float* __restrict__ h,
    const _Float16* __restrict__ Wl, const _Float16* __restrict__ Wr,
    const float* __restrict__ bias, float* __restrict__ out, int n_nodes) {
    __shared__ _Float16 sWl[HDIM * 32];
    __shared__ _Float16 sWr[HDIM * 32];

    const int t    = threadIdx.x;
    const int lane = t & 31;
    const int wave = t >> 5;
    const int m    = lane & 15;   // row (A) / column (B,D) within tile
    const int hi   = lane >> 4;   // K-half selector
    const int tile     = blockIdx.x * 8 + wave;
    const int row_base = tile * 16;
    const bool active  = (row_base < n_nodes);     // wave-uniform

    const int arow = active ? (row_base + m) : 0;
    float rdeg = 1.0f;
    if (active) rdeg = 1.0f / fmaxf(deg[arow], 1.0f);
    const float* __restrict__ aggrow = agg + (size_t)arow * K;
    const float* __restrict__ hrow   = h   + (size_t)arow * K;

    v8f acc[8] = {};

    const int KT = K / 32;
    for (int kt = 0; kt < KT; ++kt) {
        const int k0 = kt * 32;
        __syncthreads();
        // stage 128x32 slices of Wl, Wr into LDS (uint4 = 8 halves each)
        #pragma unroll
        for (int j = 0; j < 2; ++j) {
            int lin = t + j * 256;        // 0..511
            int r   = lin >> 2;           // weight row 0..127
            int c8  = (lin & 3) * 8;      // 0,8,16,24
            *(uint4*)&sWl[r * 32 + c8] = *(const uint4*)&Wl[(size_t)r * K + k0 + c8];
            *(uint4*)&sWr[r * 32 + c8] = *(const uint4*)&Wr[(size_t)r * K + k0 + c8];
        }
        __syncthreads();

        if (active) {
            // A fragments: elems 0..7 -> K = k0+hi*8.., elems 8..15 -> K = k0+16+hi*8..
            const int kb0 = k0 + hi * 8;
            const int kb1 = k0 + 16 + hi * 8;
            union { _Float16 e[16]; v16h v; } u1, u2;
            #pragma unroll
            for (int i = 0; i < 8; ++i) {
                u1.e[i]     = (_Float16)(aggrow[kb0 + i] * rdeg);
                u1.e[8 + i] = (_Float16)(aggrow[kb1 + i] * rdeg);
                u2.e[i]     = (_Float16)hrow[kb0 + i];
                u2.e[8 + i] = (_Float16)hrow[kb1 + i];
            }
            v16h a1 = u1.v, a2 = u2.v;

            #pragma unroll
            for (int nt = 0; nt < 8; ++nt) {
                const int w = nt * 16 + m;   // weight row = output column
                union { _Float16 e[16]; v16h v; } bl, br;
                #pragma unroll
                for (int i = 0; i < 8; ++i) {
                    bl.e[i]     = sWl[w * 32 + hi * 8 + i];
                    bl.e[8 + i] = sWl[w * 32 + 16 + hi * 8 + i];
                    br.e[i]     = sWr[w * 32 + hi * 8 + i];
                    br.e[8 + i] = sWr[w * 32 + 16 + hi * 8 + i];
                }
                acc[nt] = __builtin_amdgcn_wmma_f32_16x16x32_f16(
                    false, a1, false, bl.v, (short)0, acc[nt], false, false);
                acc[nt] = __builtin_amdgcn_wmma_f32_16x16x32_f16(
                    false, a2, false, br.v, (short)0, acc[nt], false, false);
            }
        }
    }

    if (active) {
        #pragma unroll
        for (int nt = 0; nt < 8; ++nt) {
            const int col = nt * 16 + m;
            const float bv = bias[col];
            #pragma unroll
            for (int r = 0; r < 8; ++r) {
                int row = row_base + hi * 8 + r;
                if (row < n_nodes) {
                    float v = acc[nt][r] + bv;
                    if (RELU) v = fmaxf(v, 0.0f);
                    out[(size_t)row * HDIM + col] = v;
                }
            }
        }
    }
}

// ---------------------------------------------------------------- head MLP
__global__ __launch_bounds__(128) void head_kernel(
    const float* __restrict__ h, const int* __restrict__ centers,
    const float* __restrict__ w1, const float* __restrict__ b1,
    const float* __restrict__ w2, const float* __restrict__ b2,
    float* __restrict__ out, int G) {
    __shared__ float pair[HDIM];
    __shared__ float red[HDIM];
    int g = blockIdx.x;
    if (g >= G) return;
    int j = threadIdx.x;
    int c = centers[g];
    pair[j] = h[(size_t)c * HDIM + j] * h[(size_t)(c + 1) * HDIM + j];
    __syncthreads();
    float s = b1[j];
    #pragma unroll 4
    for (int k = 0; k < HDIM; ++k) s += w1[j * HDIM + k] * pair[k];
    s = fmaxf(s, 0.0f);
    red[j] = s * w2[j];
    __syncthreads();
    for (int off = 64; off > 0; off >>= 1) {
        if (j < off) red[j] += red[j + off];
        __syncthreads();
    }
    if (j == 0) out[g] = red[0] + b2[0];
}

// ---------------------------------------------------------------- launcher
extern "C" void kernel_launch(void* const* d_in, const int* in_sizes, int n_in,
                              void* d_out, int out_size, void* d_ws, size_t ws_size,
                              hipStream_t stream) {
    const int N = in_sizes[0];
    const int E = in_sizes[1] / 2;
    const int G = out_size;

    const int*   z      = (const int*)d_in[0];
    const int*   esrc   = (const int*)d_in[1];
    const int*   edst   = esrc + E;
    const int*   batch  = (const int*)d_in[2];
    const float* x      = (const float*)d_in[3];
    const float* ztab   = (const float*)d_in[5];
    const float* Wl0    = (const float*)d_in[6];
    const float* bl0    = (const float*)d_in[7];
    const float* Wr0    = (const float*)d_in[8];
    const float* Wl1    = (const float*)d_in[9];
    const float* bl1    = (const float*)d_in[10];
    const float* Wr1    = (const float*)d_in[11];
    const float* Wl2    = (const float*)d_in[12];
    const float* bl2    = (const float*)d_in[13];
    const float* Wr2    = (const float*)d_in[14];
    const float* l1w    = (const float*)d_in[15];
    const float* l1b    = (const float*)d_in[16];
    const float* l2w    = (const float*)d_in[17];
    const float* l2b    = (const float*)d_in[18];

    // ---- workspace layout
    char* ws = (char*)d_ws;
    size_t off = 0;
    float* bufA = (float*)(ws + off); off += (size_t)N * 256 * 4;   // h0 / h2
    float* agg  = (float*)(ws + off); off += (size_t)N * 256 * 4;
    float* bufB = (float*)(ws + off); off += (size_t)N * HDIM * 4;  // h1 / h3
    float* deg  = (float*)(ws + off); off += (size_t)N * 4;
    _Float16* Wh = (_Float16*)(ws + off); off += 131072 * 2;        // f16 weights
    int* centers = (int*)(ws + off); off += (size_t)G * 4;

    _Float16* Wl0h = Wh;            // 128x256
    _Float16* Wr0h = Wh + 32768;    // 128x256
    _Float16* Wl1h = Wh + 65536;    // 128x128
    _Float16* Wr1h = Wh + 81920;
    _Float16* Wl2h = Wh + 98304;
    _Float16* Wr2h = Wh + 114688;

    const int T = 256;
    #define NB(n) (((n) + T - 1) / T)

    // degree (once) + centers
    zero_kernel<<<NB(N), T, 0, stream>>>(deg, N);
    degree_kernel<<<NB(E), T, 0, stream>>>(edst, deg, E);
    centers_kernel<<<NB(N), T, 0, stream>>>(batch, centers, N);

    // f32 -> f16 weights
    f32_to_f16_kernel<<<NB(32768), T, 0, stream>>>(Wl0, Wl0h, 32768);
    f32_to_f16_kernel<<<NB(32768), T, 0, stream>>>(Wr0, Wr0h, 32768);
    f32_to_f16_kernel<<<NB(16384), T, 0, stream>>>(Wl1, Wl1h, 16384);
    f32_to_f16_kernel<<<NB(16384), T, 0, stream>>>(Wr1, Wr1h, 16384);
    f32_to_f16_kernel<<<NB(16384), T, 0, stream>>>(Wl2, Wl2h, 16384);
    f32_to_f16_kernel<<<NB(16384), T, 0, stream>>>(Wr2, Wr2h, 16384);

    // h0 = [z_table[z] | x]
    embed_kernel<<<NB(N * 64), T, 0, stream>>>(z, x, ztab, bufA, N);

    const int tiles  = (N + 15) / 16;
    const int gblocks = (tiles + 7) / 8;

    // ---- layer 0 (K = 256), ReLU
    zero_kernel<<<NB(N * 256), T, 0, stream>>>(agg, N * 256);
    scatter_add_kernel<<<NB(E * 64), T, 0, stream>>>(bufA, esrc, edst, agg, E, 64);
    sage_gemm_kernel<256, true><<<gblocks, 256, 0, stream>>>(
        agg, deg, bufA, Wl0h, Wr0h, bl0, bufB, N);

    // ---- layer 1 (K = 128), ReLU
    zero_kernel<<<NB(N * 128), T, 0, stream>>>(agg, N * 128);
    scatter_add_kernel<<<NB(E * 32), T, 0, stream>>>(bufB, esrc, edst, agg, E, 32);
    sage_gemm_kernel<128, true><<<gblocks, 256, 0, stream>>>(
        agg, deg, bufB, Wl1h, Wr1h, bl1, bufA, N);

    // ---- layer 2 (K = 128), no ReLU
    zero_kernel<<<NB(N * 128), T, 0, stream>>>(agg, N * 128);
    scatter_add_kernel<<<NB(E * 32), T, 0, stream>>>(bufA, esrc, edst, agg, E, 32);
    sage_gemm_kernel<128, false><<<gblocks, 256, 0, stream>>>(
        agg, deg, bufA, Wl2h, Wr2h, bl2, bufB, N);

    // ---- head
    head_kernel<<<G, 128, 0, stream>>>(bufB, centers, l1w, l1b, l2w, l2b,
                                       (float*)d_out, G);
    #undef NB
    (void)n_in; (void)ws_size;
}
